// MatrixRebuild_29300266893702
// MI455X (gfx1250) — matrix operand order repaired and checked by reading, not compiled, not run on hardware
//
#include <hip/hip_runtime.h>
#include <hip/hip_bf16.h>
#include <stdint.h>

typedef __attribute__((ext_vector_type(16))) _Float16 v16h;
typedef __attribute__((ext_vector_type(8)))  float    v8f;
typedef unsigned long long u64;

#define SORT_BLOCK 256
#define SORT_CHUNK 2048
#define SORT_ITEMS 8

// ---------------------------------------------------------------------------
// 256-thread exclusive scan (Hillis-Steele in LDS). Returns exclusive prefix,
// writes block total to *total. Contains barriers: all threads' pre-call
// reads are complete before any thread returns.
// ---------------------------------------------------------------------------
__device__ __forceinline__ unsigned block_exscan256(unsigned val,
                                                    volatile unsigned* buf,
                                                    unsigned* total)
{
    const int t = threadIdx.x;
    buf[t] = val;
    __syncthreads();
#pragma unroll
    for (int off = 1; off < 256; off <<= 1) {
        unsigned x = (t >= off) ? buf[t - off] : 0u;
        __syncthreads();
        buf[t] += x;
        __syncthreads();
    }
    unsigned incl = buf[t];
    *total = buf[255];
    __syncthreads();
    return incl - val;
}

// ---------------------------------------------------------------------------
// K0: convert W1 (f32 [64][128], row d-major = exactly the WMMA B source
// layout we want: B[k][n] = W1[n][k]) to f16 in workspace.
// ---------------------------------------------------------------------------
__global__ void k_convert_w1(const float* __restrict__ w1,
                             _Float16* __restrict__ w1h, int n)
{
    int i = blockIdx.x * blockDim.x + threadIdx.x;
    if (i < n) w1h[i] = (_Float16)w1[i];
}

// ---------------------------------------------------------------------------
// K1: edge MLP with WMMA. One wave processes 16 edges per tile:
//   X[16x128] (gathered src|dst embeddings, f32->f16) @ W1^T[128x64]
// as 4 N-tiles x 4 K-steps of V_WMMA_F32_16X16X32_F16, bias+ReLU+W2 dot
// fused in C-fragment registers, 16-lane shuffle reduction, then the
// Gumbel-sigmoid tail. Emits packed sort key (row<<20|col) + value.
// ---------------------------------------------------------------------------
__global__ void __launch_bounds__(256)
k_edge_mlp_wmma(const float* __restrict__ emb,
                const _Float16* __restrict__ w1h,   // [64][128] f16
                const float* __restrict__ b1,       // [64]
                const float* __restrict__ w2,       // [64]
                const float* __restrict__ b2,       // [1]
                const int* __restrict__ ei,         // [2*E] src then dst
                const float* __restrict__ eps_u,    // [E]
                const float* __restrict__ rb_u,     // [E]
                u64* __restrict__ keys,
                float* __restrict__ vals,
                int E, int numTiles, int totalWaves)
{
    const int lane = threadIdx.x & 31;
    const int m    = lane & 15;   // row-in-tile for A, col for B/C
    const int hi   = lane >> 4;   // half-wave selector
    const int waveGlobal = (blockIdx.x * blockDim.x + threadIdx.x) >> 5;

    // --- Preload all 16 B fragments of W1^T into registers (amortized over
    //     the grid-stride tile loop). B lane mapping: N = m, 16 consecutive
    //     K halves starting at kt*32 + hi*16.  32B/lane -> 2x b128 loads.
    union BU { v16h v; uint4 q[2]; };
    v16h bf[4][4];
#pragma unroll
    for (int kt = 0; kt < 4; ++kt) {
#pragma unroll
        for (int nt = 0; nt < 4; ++nt) {
            const int row  = nt * 16 + m;          // output dim d
            const int kcol = kt * 32 + hi * 16;    // K half index
            const uint4* p = (const uint4*)(w1h + row * 128 + kcol);
            BU bu;
            bu.q[0] = p[0];
            bu.q[1] = p[1];
            bf[kt][nt] = bu.v;
        }
    }

    const float b2v = b2[0];

    for (int tile = waveGlobal; tile < numTiles; tile += totalWaves) {
        const int e   = tile * 16 + m;
        const bool eok = (e < E);
        const int srcI = eok ? ei[e]     : 0;
        const int dstI = eok ? ei[E + e] : 0;

        // --- A fragments straight from global (L2-resident table), matching
        //     the B layout: halves j -> K = 16*hi + j (per 32-wide K tile).
        //     kt 0,1 read the src row (X cols 0..63), kt 2,3 the dst row.
        union AU { v16h v; _Float16 h[16]; };
        v16h af[4];
#pragma unroll
        for (int kt = 0; kt < 4; ++kt) {
            const float* rp = emb + (size_t)((kt < 2) ? srcI : dstI) * 64;
            const int kb = (kt & 1) * 32 + hi * 16;
            float4 f0 = *(const float4*)(rp + kb);
            float4 f1 = *(const float4*)(rp + kb + 4);
            float4 g0 = *(const float4*)(rp + kb + 8);
            float4 g1 = *(const float4*)(rp + kb + 12);
            AU a;
            a.h[0]  = (_Float16)f0.x; a.h[1]  = (_Float16)f0.y;
            a.h[2]  = (_Float16)f0.z; a.h[3]  = (_Float16)f0.w;
            a.h[4]  = (_Float16)f1.x; a.h[5]  = (_Float16)f1.y;
            a.h[6]  = (_Float16)f1.z; a.h[7]  = (_Float16)f1.w;
            a.h[8]  = (_Float16)g0.x; a.h[9]  = (_Float16)g0.y;
            a.h[10] = (_Float16)g0.z; a.h[11] = (_Float16)g0.w;
            a.h[12] = (_Float16)g1.x; a.h[13] = (_Float16)g1.y;
            a.h[14] = (_Float16)g1.z; a.h[15] = (_Float16)g1.w;
            af[kt] = a.v;
        }

        float acc[8];
#pragma unroll
        for (int r = 0; r < 8; ++r) acc[r] = 0.f;

#pragma unroll
        for (int nt = 0; nt < 4; ++nt) {
            v8f c = {0.f, 0.f, 0.f, 0.f, 0.f, 0.f, 0.f, 0.f};
#pragma unroll
            for (int kt = 0; kt < 4; ++kt) {
                c = __builtin_amdgcn_wmma_f32_16x16x32_f16(
                        false, af[kt], false, bf[kt][nt],
                        (short)0, c, false, false);
            }
            // Fused bias + ReLU + W2 dot: lane holds column d = nt*16+m for
            // rows r+8*hi; never materialize h.
            const int   d   = nt * 16 + m;
            const float b1v = b1[d];
            const float w2v = w2[d];
#pragma unroll
            for (int r = 0; r < 8; ++r) {
                float t = c[r] + b1v;
                t = t > 0.f ? t : 0.f;
                acc[r] = fmaf(t, w2v, acc[r]);
            }
        }

        // Reduce the 64-column dot across the 16 lanes of each half-wave.
#pragma unroll
        for (int r = 0; r < 8; ++r) {
            float s = acc[r];
            s += __shfl_xor(s, 1, 32);
            s += __shfl_xor(s, 2, 32);
            s += __shfl_xor(s, 4, 32);
            s += __shfl_xor(s, 8, 32);
            acc[r] = s;
        }

        // Gumbel-sigmoid tail on lanes 0 and 16 (8 edges each).
        if (m == 0) {
#pragma unroll
            for (int r = 0; r < 8; ++r) {
                const int e2 = tile * 16 + hi * 8 + r;
                if (e2 >= E) continue;
                const float logit = acc[r] + b2v;
                const float epsu  = eps_u[e2];
                const float eps   = fmaf(-0.9998f, epsu, 0.9999f);
                const float gate  = logf(eps) - log1pf(-eps) + logit;
                const float w     = 1.f / (1.f + expf(-gate));
                const float att   = fminf(fmaxf(w, 0.01f), 0.99f);
                const float u     = fminf(fmaxf(rb_u[e2], 1e-7f), 1.f - 1e-7f);
                const float s     = (logf(att) - log1pf(-att) +
                                     logf(u)   - log1pf(-u)) * (1.f / 0.9f);
                const float sample = 1.f / (1.f + expf(-s)); // mask==1 always
                const unsigned r0 = (unsigned)ei[e2];
                const unsigned c0 = (unsigned)ei[E + e2];
                keys[e2] = ((u64)r0 << 20) | (u64)c0;  // lexsort(rows, cols)
                vals[e2] = sample;
            }
        }
    }
}

// ---------------------------------------------------------------------------
// K2: append graph_random entries (value 0.05).
// ---------------------------------------------------------------------------
__global__ void k_fill_random(const int* __restrict__ ridx,
                              u64* __restrict__ keys, float* __restrict__ vals,
                              int R, int E)
{
    int i = blockIdx.x * blockDim.x + threadIdx.x;
    if (i >= R) return;
    const unsigned r = (unsigned)ridx[i];
    const unsigned c = (unsigned)ridx[R + i];
    keys[E + i] = ((u64)r << 20) | (u64)c;
    vals[E + i] = 0.05f;
}

// ---------------------------------------------------------------------------
// Radix sort, 5 passes x 8-bit digits over 39-bit packed keys.
// ---------------------------------------------------------------------------
__global__ void __launch_bounds__(SORT_BLOCK)
k_radix_hist(const u64* __restrict__ keys, unsigned* __restrict__ hist,
             int M, int numBlocks, int shift)
{
    __shared__ unsigned lh[256];
    lh[threadIdx.x] = 0;
    __syncthreads();
    const int base = blockIdx.x * SORT_CHUNK;
    for (int i = threadIdx.x; i < SORT_CHUNK; i += SORT_BLOCK) {
        const int g = base + i;
        if (g < M) {
            const unsigned d = (unsigned)((keys[g] >> shift) & 255ull);
            atomicAdd(&lh[d], 1u);
        }
    }
    __syncthreads();
    // digit-major so a flat exclusive scan yields global (digit, block) bases
    hist[threadIdx.x * numBlocks + blockIdx.x] = lh[threadIdx.x];
}

__global__ void __launch_bounds__(256)
k_scan(unsigned* __restrict__ data, int n)
{
    __shared__ unsigned buf[256];
    __shared__ unsigned carry_s;
    if (threadIdx.x == 0) carry_s = 0;
    __syncthreads();
    for (int base = 0; base < n; base += 256) {
        const int i = base + threadIdx.x;
        unsigned v = (i < n) ? data[i] : 0u;
        unsigned tot;
        const unsigned ex = block_exscan256(v, buf, &tot);
        const unsigned carry = carry_s;
        if (i < n) data[i] = ex + carry;
        __syncthreads();
        if (threadIdx.x == 0) carry_s += tot;
        __syncthreads();
    }
}

__global__ void __launch_bounds__(SORT_BLOCK)
k_radix_scatter(const u64* __restrict__ keysIn, const float* __restrict__ valsIn,
                u64* __restrict__ keysOut, float* __restrict__ valsOut,
                const unsigned* __restrict__ scanned,
                int M, int numBlocks, int shift)
{
    __shared__ u64      skey[SORT_CHUNK];
    __shared__ float    sval[SORT_CHUNK];
    __shared__ unsigned sbuf[256];
    __shared__ unsigned lhist[256];
    __shared__ unsigned dstart[256];

    const int t    = threadIdx.x;
    const int blk  = blockIdx.x;
    const int base = blk * SORT_CHUNK;

    for (int i = t; i < SORT_CHUNK; i += SORT_BLOCK) {
        const int g = base + i;
        skey[i] = (g < M) ? keysIn[g] : ~0ull;   // sentinel sorts last
        sval[i] = (g < M) ? valsIn[g] : 0.f;
    }
    __syncthreads();

    // Stable local sort of the current 8-bit digit via 8 one-bit splits.
    for (int bit = 0; bit < 8; ++bit) {
        u64 k[SORT_ITEMS]; float v[SORT_ITEMS]; int b[SORT_ITEMS];
        unsigned zc = 0;
#pragma unroll
        for (int i = 0; i < SORT_ITEMS; ++i) {
            k[i] = skey[t * SORT_ITEMS + i];
            v[i] = sval[t * SORT_ITEMS + i];
            b[i] = (int)((k[i] >> (shift + bit)) & 1ull);
            zc  += (unsigned)(1 - b[i]);
        }
        unsigned Z;
        const unsigned ex = block_exscan256(zc, sbuf, &Z); // barriers inside
        int zrun = (int)ex;
        int orun = (int)Z + t * SORT_ITEMS - (int)ex;
#pragma unroll
        for (int i = 0; i < SORT_ITEMS; ++i) {
            const int pos = b[i] ? orun++ : zrun++;
            skey[pos] = k[i];
            sval[pos] = v[i];
        }
        __syncthreads();
    }

    // Local digit histogram + exclusive scan -> run starts.
    lhist[t] = 0;
    __syncthreads();
#pragma unroll
    for (int i = 0; i < SORT_ITEMS; ++i) {
        const unsigned d = (unsigned)((skey[t * SORT_ITEMS + i] >> shift) & 255ull);
        atomicAdd(&lhist[d], 1u);
    }
    __syncthreads();
    unsigned tot;
    const unsigned ex = block_exscan256(lhist[t], sbuf, &tot);
    dstart[t] = ex;
    __syncthreads();

    // Scatter: globalBase(digit, block) + local rank within digit run.
#pragma unroll
    for (int i = 0; i < SORT_ITEMS; ++i) {
        const int p = t * SORT_ITEMS + i;
        const u64 k = skey[p];
        if (k == ~0ull) continue;                // padding
        const unsigned d = (unsigned)((k >> shift) & 255ull);
        const unsigned g = scanned[d * numBlocks + blk] + (unsigned)p - dstart[d];
        keysOut[g] = k;
        valsOut[g] = sval[p];
    }
}

// ---------------------------------------------------------------------------
// K3: coalesce. change at first occurrence gets run sum, duplicates get 0.
// d_out = [rows (M) | cols (M) | vals (M)] as f32.
// ---------------------------------------------------------------------------
__global__ void k_coalesce(const u64* __restrict__ keys,
                           const float* __restrict__ vals,
                           float* __restrict__ out, int M)
{
    const int i = blockIdx.x * blockDim.x + threadIdx.x;
    if (i >= M) return;
    const u64 k = keys[i];
    out[i]     = (float)(unsigned)(k >> 20);        // row
    out[M + i] = (float)(unsigned)(k & 0xFFFFFull); // col
    float v = 0.f;
    const bool first = (i == 0) || (keys[i - 1] != k);
    if (first) {
        v = vals[i];
        int j = i + 1;
        while (j < M && keys[j] == k) { v += vals[j]; ++j; }
    }
    out[2 * M + i] = v;
}

// ---------------------------------------------------------------------------
extern "C" void kernel_launch(void* const* d_in, const int* in_sizes, int n_in,
                              void* d_out, int out_size, void* d_ws, size_t ws_size,
                              hipStream_t stream)
{
    (void)n_in; (void)out_size; (void)ws_size;

    const float* emb  = (const float*)d_in[0];
    const float* W1   = (const float*)d_in[1];
    const float* b1   = (const float*)d_in[2];
    const float* W2   = (const float*)d_in[3];
    const float* b2   = (const float*)d_in[4];
    const int*   ei   = (const int*)d_in[5];
    const int*   ridx = (const int*)d_in[6];
    const float* epsu = (const float*)d_in[7];
    const float* rbu  = (const float*)d_in[8];

    const int E = in_sizes[5] / 2;
    const int R = in_sizes[6] / 2;
    const int M = E + R;

    auto align256 = [](size_t x) { return (x + 255) & ~(size_t)255; };
    char* ws = (char*)d_ws;
    size_t off = 0;
    _Float16* w1h = (_Float16*)(ws + off); off += align256(64 * 128 * sizeof(_Float16));
    u64*   keys0 = (u64*)(ws + off);   off += align256((size_t)M * sizeof(u64));
    u64*   keys1 = (u64*)(ws + off);   off += align256((size_t)M * sizeof(u64));
    float* vals0 = (float*)(ws + off); off += align256((size_t)M * sizeof(float));
    float* vals1 = (float*)(ws + off); off += align256((size_t)M * sizeof(float));
    const int numBlocks = (M + SORT_CHUNK - 1) / SORT_CHUNK;
    unsigned* hist = (unsigned*)(ws + off); off += align256((size_t)numBlocks * 256 * sizeof(unsigned));

    // K0: weight conversion
    k_convert_w1<<<(64 * 128 + 255) / 256, 256, 0, stream>>>(W1, w1h, 64 * 128);

    // K1: edge MLP (WMMA) -> keys/vals [0, E)
    const int numTiles   = (E + 15) / 16;
    const int blocks     = 512;                // 512 blocks x 8 waves
    const int totalWaves = blocks * (256 / 32);
    k_edge_mlp_wmma<<<blocks, 256, 0, stream>>>(emb, w1h, b1, W2, b2, ei,
                                                epsu, rbu, keys0, vals0,
                                                E, numTiles, totalWaves);

    // K2: random entries -> keys/vals [E, M)
    k_fill_random<<<(R + 255) / 256, 256, 0, stream>>>(ridx, keys0, vals0, R, E);

    // Radix sort: 5 x 8-bit passes over 39-bit keys (ping-pong).
    u64*   kin = keys0; u64*   kout = keys1;
    float* vin = vals0; float* vout = vals1;
    for (int pass = 0; pass < 5; ++pass) {
        const int shift = pass * 8;
        k_radix_hist<<<numBlocks, SORT_BLOCK, 0, stream>>>(kin, hist, M, numBlocks, shift);
        k_scan<<<1, 256, 0, stream>>>(hist, numBlocks * 256);
        k_radix_scatter<<<numBlocks, SORT_BLOCK, 0, stream>>>(kin, vin, kout, vout,
                                                              hist, M, numBlocks, shift);
        u64* tk = kin; kin = kout; kout = tk;
        float* tv = vin; vin = vout; vout = tv;
    }

    // K3: coalesce + emit (rows, cols, vals)
    k_coalesce<<<(M + 255) / 256, 256, 0, stream>>>(kin, vin, (float*)d_out, M);
}
